// ModuleCorrelation_58402965291357
// MI455X (gfx1250) — matrix operand order repaired and checked by reading, not compiled, not run on hardware
//
#include <hip/hip_runtime.h>

// ---------------------------------------------------------------------------
// FlowNet correlation (pad=20, max_disp=20, stride2=2, K=1) on gfx1250 WMMA.
// out[b, dy*21+t, h, w] = (1/256) * sum_c first[b,c,h,w]*second[b,c,h+2dy-20,w+2t-20]
//
// Parity-split banded GEMM: j = w + 2t has parity(w).  With w = 2r+p, j = 2q+p,
// q = r+t:   G_p[r,q] = sum_c A[2r+p,c] * Bp[c,2q+p],  out(2r+p, t) = G_p[r, r+t].
// A 16-row tile r0 needs q in [r0, r0+36) -> exactly 3 aligned 16-wide N-tiles
// (2.29x redundancy vs 3.05x for the mixed-parity mapping).
// ---------------------------------------------------------------------------

typedef __attribute__((ext_vector_type(16))) __bf16 v16bf;
typedef __attribute__((ext_vector_type(8)))  float  v8f;

#define BATCH   8
#define CH      256
#define HH      96
#define WW      128
#define GRIDW   21            // displacements per axis
#define NCH     441
#define JW      192           // staged padded B width: j in [0,192), j = w+20+2t domain
#define QW      96            // half-index columns per parity
#define PITCH   264           // halves per LDS row (256 + 8 pad; 528B row = 4-bank lane stride)

__device__ __forceinline__ unsigned short f2bf(float f) {
    union { float f; unsigned u; } x; x.f = f;
    unsigned r = x.u + 0x7FFFu + ((x.u >> 16) & 1u);   // round-to-nearest-even
    return (unsigned short)(r >> 16);
}

union FragU { uint4 q[2]; v16bf v; };

// A fragment: halves [p..p+7] and [p+16..p+23]  (16-bit 16x32 A VGPR layout)
__device__ __forceinline__ v16bf load_fragA(const unsigned short* p) {
    FragU f;
    f.q[0] = *(const uint4*)(p);
    f.q[1] = *(const uint4*)(p + 16);
    return f.v;
}
// B fragment: halves [p..p+7] and [p+16..p+23]  (same K order as A)
__device__ __forceinline__ v16bf load_fragB(const unsigned short* p) {
    FragU f;
    f.q[0] = *(const uint4*)(p);
    f.q[1] = *(const uint4*)(p + 16);
    return f.v;
}

__global__ __launch_bounds__(256)
void ModuleCorrelation_58402965291357_kernel(const float* __restrict__ first,
                                             const float* __restrict__ second,
                                             float* __restrict__ out)
{
    extern __shared__ unsigned short lds[];
    unsigned short* ldsA = lds;                        // [2][64][PITCH]: row = (w&1)*64 + (w>>1)
    unsigned short* ldsB = lds + WW * PITCH;           // [2][96][PITCH]: row = (j&1)*96 + (j>>1)

    const int h    = blockIdx.x;
    const int b    = blockIdx.y;
    const int tid  = threadIdx.x;
    const int lane = tid & 31;
    const int wave = tid >> 5;

    // ---- stage A = first[b,:,h,:] as bf16, parity-deinterleaved [p][r][c] ----
    const float* fbase = first + ((size_t)(b * CH) * HH + h) * WW;
    for (int idx = tid; idx < CH * WW; idx += 256) {
        int c = idx >> 7;
        int w = idx & (WW - 1);
        int row = (w & 1) * 64 + (w >> 1);
        ldsA[row * PITCH + c] = f2bf(fbase[(size_t)c * (HH * WW) + w]);
    }

    // wave -> (parity, M-tile): 8 waves cover 2 parities x 4 tiles of 16 half-rows
    const int p      = wave & 1;
    const int r0     = (wave >> 1) * 16;
    const int rr     = lane & 15;
    const int kselA  = (lane >> 4) * 8;    // lanes 0-15: K {0..7,16..23}; 16-31: {8..15,24..31}
    const int nn     = lane & 15;
    const int kselB  = (lane >> 4) * 8;    // lanes 0-15: K {0..7,16..23}; 16-31: {8..15,24..31}
    const float scale = 1.0f / 256.0f;

    const unsigned short* aBase = &ldsA[(p * 64 + r0 + rr) * PITCH + kselA];
    const unsigned short* bBase0 = &ldsB[(p * QW + r0      + nn) * PITCH + kselB];
    const unsigned short* bBase1 = &ldsB[(p * QW + r0 + 16 + nn) * PITCH + kselB];
    const unsigned short* bBase2 = &ldsB[(p * QW + r0 + 32 + nn) * PITCH + kselB];

    for (int dy = 0; dy < GRIDW; ++dy) {
        const int  hp    = h + 2 * dy - 20;
        const bool inRow = (hp >= 0) && (hp < HH);

        __syncthreads();                               // previous-iter readers done with ldsB
        if (inRow) {
            const float* sbase = second + ((size_t)(b * CH) * HH + hp) * WW;
            for (int idx = tid; idx < CH * JW; idx += 256) {
                int c = idx / JW;
                int j = idx - c * JW;
                int w = j - 20;
                float v = (w >= 0 && w < WW) ? sbase[(size_t)c * (HH * WW) + w] : 0.0f;
                int row = (j & 1) * QW + (j >> 1);
                ldsB[row * PITCH + c] = f2bf(v);
            }
            // prefetch next displaced row into cache while this dy computes
            int hpn = hp + 2;
            if (hpn < HH) {
                const char* pf = (const char*)(second + ((size_t)(b * CH) * HH + hpn) * WW);
                #pragma unroll
                for (int i = 0; i < 8; ++i)
                    __builtin_prefetch(pf + ((size_t)tid + i * 256) * 64, 0, 3);
            }
        }
        __syncthreads();

        v8f acc0 = (v8f){0.f,0.f,0.f,0.f,0.f,0.f,0.f,0.f};
        v8f acc1 = acc0, acc2 = acc0;

        if (inRow) {
            // software-pipelined fragments: next k-step's DS loads issue before
            // the current k-step's WMMAs consume, so waits retire early.
            v16bf a  = load_fragA(aBase);
            v16bf b0 = load_fragB(bBase0);
            v16bf b1 = load_fragB(bBase1);
            v16bf b2 = load_fragB(bBase2);
            #pragma unroll
            for (int k0 = 0; k0 < CH; k0 += 32) {
                int kn = (k0 + 32 < CH) ? (k0 + 32) : (CH - 32);   // dup-load on last iter
                v16bf an = load_fragA(aBase + kn);
                v16bf c0 = load_fragB(bBase0 + kn);
                v16bf c1 = load_fragB(bBase1 + kn);
                v16bf c2 = load_fragB(bBase2 + kn);
                acc0 = __builtin_amdgcn_wmma_f32_16x16x32_bf16(false, a, false, b0,
                                                               (short)0, acc0, false, false);
                acc1 = __builtin_amdgcn_wmma_f32_16x16x32_bf16(false, a, false, b1,
                                                               (short)0, acc1, false, false);
                acc2 = __builtin_amdgcn_wmma_f32_16x16x32_bf16(false, a, false, b2,
                                                               (short)0, acc2, false, false);
                a = an; b0 = c0; b1 = c1; b2 = c2;
            }
        }

        // ---- extract band: t = 16n + nn - rr(row);  w = 2(r0+row)+p; unique writer ----
        #pragma unroll
        for (int n = 0; n < 3; ++n) {
            const v8f acc = (n == 0) ? acc0 : (n == 1) ? acc1 : acc2;
            #pragma unroll
            for (int v = 0; v < 8; ++v) {
                int row = v + ((lane >> 4) << 3);      // D layout: M = v (+8 for upper lanes)
                int t   = 16 * n + nn - row;
                if (t >= 0 && t <= 40 - 20) {          // t in [0, 20]
                    int w  = 2 * (r0 + row) + p;
                    int ch = dy * GRIDW + t;
                    out[(((size_t)b * NCH + ch) * HH + h) * WW + w] = acc[v] * scale;
                }
            }
        }
    }
}

extern "C" void kernel_launch(void* const* d_in, const int* in_sizes, int n_in,
                              void* d_out, int out_size, void* d_ws, size_t ws_size,
                              hipStream_t stream) {
    (void)in_sizes; (void)n_in; (void)out_size; (void)d_ws; (void)ws_size;
    const float* first  = (const float*)d_in[0];
    const float* second = (const float*)d_in[1];
    float*       out    = (float*)d_out;

    const size_t shmem = (size_t)(WW * PITCH + JW * PITCH) * sizeof(unsigned short); // ~165 KB
    static bool attrSet = false;   // idempotent host-side attribute, not a stream op
    if (!attrSet) {
        hipFuncSetAttribute((const void*)ModuleCorrelation_58402965291357_kernel,
                            hipFuncAttributeMaxDynamicSharedMemorySize, (int)shmem);
        attrSet = true;
    }

    dim3 grid(HH, BATCH);   // 96 x 8 = 768 workgroups, one per (h, b)
    dim3 block(256);        // 8 waves: one (parity, M-tile) per wave, 3 N-tiles each
    ModuleCorrelation_58402965291357_kernel<<<grid, block, shmem, stream>>>(first, second, out);
}
